// KShape_51977694216600
// MI455X (gfx1250) — matrix operand order repaired and checked by reading, not compiled, not run on hardware
//
#include <hip/hip_runtime.h>

// ---------------------------------------------------------------------------
// Problem constants (from reference): B=256, K=16, SZ=512, D=8
// Flattened reduction length R = SZ*D = 4096, lags L = 2*SZ-1 = 1023.
// Zero-padded x rows: TPAD = SZ + 2*(SZ-1) = 1534 time steps.
// ---------------------------------------------------------------------------
#define NB      256
#define NK      16
#define NSZ     512
#define ND      8
#define NR      4096          // SZ*D
#define NLAG    1023          // 2*SZ-1
#define TPAD    1534          // padded time steps per row
#define ROWE    (TPAD * ND)   // 12272 bf16 elems per padded row
#define NCHUNK  128           // NR / 32 K-chunks per WMMA dot
#define LAGS_WG 64            // lags per workgroup (8 waves x 8 lags)
#define TWIN    (511 + LAGS_WG)          // 575 time steps of window in LDS
#define LDS_BYTES (16 * TWIN * ND * 2)   // 147200 B (<320KB WGP LDS)

typedef __attribute__((ext_vector_type(16))) __bf16      v16bf;
typedef __attribute__((ext_vector_type(8)))  float       v8f;
typedef __attribute__((ext_vector_type(8)))  unsigned int v8u;

// float -> bf16 (round to nearest even), bit-level
__device__ __forceinline__ unsigned short f2bf(float f) {
    unsigned int u = __float_as_uint(f);
    unsigned int r = u + 0x7FFFu + ((u >> 16) & 1u);
    return (unsigned short)(r >> 16);
}

// monotonic float atomic max (handles negatives; init value must be -inf)
__device__ __forceinline__ void atomicMaxFloat(float* addr, float value) {
    if (value >= 0.0f) atomicMax((int*)addr, __float_as_int(value));
    else               atomicMin((unsigned int*)addr, __float_as_uint(value));
}

// ---------------------------------------------------------------------------
// ① zero-padded bf16 copy of x: xpad[row][t'][d], t' in [0,1534)
//    xpad[t'] = bf16(x[t'-511]) for t' in [511,1023), else 0
// ---------------------------------------------------------------------------
__global__ void prep_xpad(const float* __restrict__ x, unsigned short* __restrict__ xpad) {
    int idx = blockIdx.x * blockDim.x + threadIdx.x;   // NB*ROWE total
    if (idx >= NB * ROWE) return;
    int row = idx / ROWE;
    int rem = idx - row * ROWE;
    int tp  = rem >> 3;
    int d   = rem & 7;
    float v = 0.0f;
    if (tp >= 511 && tp < 511 + NSZ) v = x[row * NR + (tp - 511) * ND + d];
    xpad[idx] = f2bf(v);
}

// ---------------------------------------------------------------------------
// ② pack centers into the exact bf16 WMMA-B (32x16) lane layout:
//    bpack[c0*256 + L*8 + v] = {bf16(B[r+1,n]) , bf16(B[r,n])}
//    with n=L&15, r = c0*32 + (v>>2)*16 + (L>>4)*8 + 2(v&3), B[r,n] = centers[n, r/8, r%8]
// ---------------------------------------------------------------------------
__global__ void prep_bpack(const float* __restrict__ centers, unsigned int* __restrict__ bpack) {
    int i = blockIdx.x * blockDim.x + threadIdx.x;     // 32768 total dwords
    if (i >= NCHUNK * 256) return;
    int c0 = i >> 8;
    int L  = (i >> 3) & 31;
    int v  = i & 7;
    int n  = L & 15;
    int r  = c0 * 32 + ((v >> 2) << 4) + ((L >> 4) << 3) + 2 * (v & 3);
    float f0 = centers[n * NR + r];
    float f1 = centers[n * NR + r + 1];
    bpack[i] = ((unsigned int)f2bf(f1) << 16) | (unsigned int)f2bf(f0);
}

// ---------------------------------------------------------------------------
// ③ per-row Frobenius norms of x, and init dists to -inf
// ---------------------------------------------------------------------------
__global__ __launch_bounds__(256) void prep_norm_init(const float* __restrict__ x,
                                                      float* __restrict__ norm1,
                                                      float* __restrict__ dists) {
    __shared__ float red[256];
    int b = blockIdx.x;
    float s = 0.0f;
    for (int i = threadIdx.x; i < NR; i += 256) {
        float v = x[b * NR + i];
        s += v * v;
    }
    red[threadIdx.x] = s;
    __syncthreads();
    for (int off = 128; off > 0; off >>= 1) {
        if (threadIdx.x < off) red[threadIdx.x] += red[threadIdx.x + off];
        __syncthreads();
    }
    if (threadIdx.x == 0) norm1[b] = sqrtf(red[0]);
    if (threadIdx.x < NK) dists[b * NK + threadIdx.x] = -__builtin_inff();
}

// ---------------------------------------------------------------------------
// ④ WMMA correlation: grid.x = 16 b-tiles, grid.y = 16 lag-chunks.
//    256 threads = 8 wave32; wave w owns lags [lag0+8w, lag0+8w+8).
//    Chunk-outer loop: B loaded once per chunk (reused by 8 lag-WMMAs),
//    A shared via 10 overlapping 16B LDS segments per chunk.
// ---------------------------------------------------------------------------
extern __shared__ __align__(16) unsigned char smem[];

__global__ __launch_bounds__(256) void corr_wmma(const unsigned short* __restrict__ xpad,
                                                 const unsigned int*  __restrict__ bpack,
                                                 const float*         __restrict__ norm1,
                                                 float*               __restrict__ dists) {
    const int btile = blockIdx.x;          // 0..15
    const int lag0  = blockIdx.y * LAGS_WG;
    const int tid   = threadIdx.x;
    const int lane  = tid & 31;
    const int wv    = tid >> 5;

    // --- cooperative LDS fill: 16 rows x TWIN t-steps x 8 d (bf16), 16B units
    for (int u = tid; u < 16 * TWIN; u += 256) {
        int row = u / TWIN;
        int tw  = u - row * TWIN;
        int tp  = lag0 + tw;
        uint4 v;
        if (tp < TPAD) {
            v = *reinterpret_cast<const uint4*>(xpad + (size_t)(btile * 16 + row) * ROWE
                                                     + (size_t)tp * ND);
        } else {
            v = make_uint4(0u, 0u, 0u, 0u);   // beyond padded row: zeros
        }
        reinterpret_cast<uint4*>(smem)[(size_t)row * TWIN + tw] = v;
    }
    __syncthreads();

    const int h = lane >> 4;       // lane half
    const int n = lane & 15;       // k column (C/D: N=lane&15) and A row

    // per-lane inverse denominators for its 8 output rows (m = 8h + i)
    float inv[8];
#pragma unroll
    for (int i = 0; i < 8; ++i) {
        float nm = norm1[btile * 16 + h * 8 + i];
        inv[i] = (n == 0) ? 0.0f : 1.0f / (nm * (float)n);
    }

    // 8 lag accumulators (one 16x16 D tile per lag handled by this wave)
    v8f acc[8];
#pragma unroll
    for (int j = 0; j < 8; ++j) acc[j] = (v8f){0, 0, 0, 0, 0, 0, 0, 0};

    const unsigned int rowBase = (unsigned int)n * (TWIN * ND);     // A row = lane&15
    const unsigned int segbase = rowBase + (unsigned int)(wv * 8) * ND + (unsigned int)h * 8;
    const unsigned int* bp = bpack + lane * 8;

    for (int c0 = 0; c0 < NCHUNK; ++c0) {
        // B (32x16 bf16): one load per chunk, reused by all 8 lag-WMMAs
        uint4 bv0 = *reinterpret_cast<const uint4*>(bp + (size_t)c0 * 256);
        uint4 bv1 = *reinterpret_cast<const uint4*>(bp + (size_t)c0 * 256 + 4);
        v8u bu = {bv0.x, bv0.y, bv0.z, bv0.w, bv1.x, bv1.y, bv1.z, bv1.w};
        v16bf b = __builtin_bit_cast(v16bf, bu);

        // 10 overlapping 16B A-segments cover the A operands of all 8 lags:
        // lag j uses segments m=j and m=j+2
        uint4 seg[10];
#pragma unroll
        for (int m = 0; m < 10; ++m) {
            unsigned int e = segbase + (unsigned int)c0 * 32 + (unsigned int)m * 8;
            seg[m] = *reinterpret_cast<const uint4*>(smem + (size_t)e * 2);
        }

#pragma unroll
        for (int j = 0; j < 8; ++j) {
            v8u au = {seg[j].x,     seg[j].y,     seg[j].z,     seg[j].w,
                      seg[j + 2].x, seg[j + 2].y, seg[j + 2].z, seg[j + 2].w};
            v16bf a = __builtin_bit_cast(v16bf, au);
            acc[j] = __builtin_amdgcn_wmma_f32_16x16x32_bf16(
                         false, a, false, b, (short)0, acc[j], false, false);
        }
    }

    // max over this wave's lags (skip the single invalid lag li==1023)
    float mx[8];
#pragma unroll
    for (int i = 0; i < 8; ++i) mx[i] = -__builtin_inff();
#pragma unroll
    for (int j = 0; j < 8; ++j) {
        if (lag0 + wv * 8 + j >= NLAG) break;   // wave-uniform; after all WMMAs
#pragma unroll
        for (int i = 0; i < 8; ++i) mx[i] = fmaxf(mx[i], acc[j][i]);
    }

    // scale (inv >= 0 commutes with max; k==0 -> inv=0 -> exact 0) and merge
#pragma unroll
    for (int i = 0; i < 8; ++i) {
        float v = inv[i] * mx[i];
        atomicMaxFloat(dists + (btile * 16 + h * 8 + i) * NK + n, v);
    }
}

// ---------------------------------------------------------------------------
// ⑤ labels[b] = argmin_k (1 - dists[b,k]) (first occurrence), stored as float
// ---------------------------------------------------------------------------
__global__ void labels_kernel(const float* __restrict__ dists, float* __restrict__ labels) {
    int b = blockIdx.x * blockDim.x + threadIdx.x;
    if (b >= NB) return;
    float best = 1.0f - dists[b * NK];
    int bi = 0;
    for (int k = 1; k < NK; ++k) {
        float v = 1.0f - dists[b * NK + k];
        if (v < best) { best = v; bi = k; }
    }
    labels[b] = (float)bi;
}

// ---------------------------------------------------------------------------
// launch
// ---------------------------------------------------------------------------
extern "C" void kernel_launch(void* const* d_in, const int* in_sizes, int n_in,
                              void* d_out, int out_size, void* d_ws, size_t ws_size,
                              hipStream_t stream) {
    const float* x       = (const float*)d_in[0];   // [256,512,8]
    const float* centers = (const float*)d_in[1];   // [16,512,8]

    // d_out layout: labels[256] then dists[256*16], all float32
    float* labels = (float*)d_out;
    float* dists  = (float*)d_out + NB;

    // workspace layout
    char* ws = (char*)d_ws;
    unsigned short* xpad  = (unsigned short*)(ws);                          // 6,283,264 B
    unsigned int*   bpack = (unsigned int*)(ws + (size_t)NB * ROWE * 2);    // 131,072 B
    float*          norm1 = (float*)(ws + (size_t)NB * ROWE * 2 + (size_t)NCHUNK * 256 * 4);

    // ① pad + bf16-convert x
    {
        int total = NB * ROWE;
        prep_xpad<<<(total + 255) / 256, 256, 0, stream>>>(x, xpad);
    }
    // ② swizzle centers into WMMA-B layout
    prep_bpack<<<NCHUNK, 256, 0, stream>>>(centers, bpack);
    // ③ norms + dists init
    prep_norm_init<<<NB, 256, 0, stream>>>(x, norm1, dists);
    // ④ WMMA correlation over all lags
    corr_wmma<<<dim3(16, (NLAG + LAGS_WG - 1) / LAGS_WG), 256, LDS_BYTES, stream>>>(
        xpad, bpack, norm1, dists);
    // ⑤ argmin(1 - dists)
    labels_kernel<<<1, 256, 0, stream>>>(dists, labels);
}
